// PerformerAttention_49091476193516
// MI455X (gfx1250) — matrix operand order repaired and checked by reading, not compiled, not run on hardware
//
#include <hip/hip_runtime.h>
#include <stdint.h>

// ---- problem constants ----
#define BB   32
#define NN   1025
#define DIMF 768
#define HH_  12
#define DD   64
#define BN   (BB*NN)       // 32800 (multiple of 32)
#define BH   (BB*HH_)      // 384
#define NP   1040          // N padded to multiple of 16 (65 tiles)
#define G3   (3*HH_*DD)    // 2304
#define GQK  (2*HH_*DD)    // 1536
#define HD   (HH_*DD)      // 768

typedef __attribute__((ext_vector_type(16))) __bf16 bf16x16;
typedef __attribute__((ext_vector_type(8)))  float  f32x8;
typedef unsigned short u16;
typedef unsigned int   u32;

union FragU { bf16x16 v; uint4 q[2]; };
union Pack8 { u16 h[8]; uint4 q; };

__device__ __forceinline__ u16 f2bf(float f) {
  union { float f; u32 u; } x; x.f = f;
  u32 u = x.u;
  u32 r = u + 0x7FFFu + ((u >> 16) & 1u);   // round-to-nearest-even
  return (u16)(r >> 16);
}
__device__ __forceinline__ float bf2f(u16 h) {
  union { u32 u; float f; } x; x.u = ((u32)h) << 16;
  return x.f;
}

// A fragment: row-major [row][k], 16x32 tile. lane%16 = M row.
// per-lane elems 0..7 at k0+(lane/16)*8, elems 8..15 at +16.
__device__ __forceinline__ bf16x16 loadA(const u16* base, int ld, int row, int k0, int lane) {
  const u16* p = base + (size_t)row * ld + k0 + ((lane >> 4) << 3);
  FragU f;
  f.q[0] = *(const uint4*)(p);
  f.q[1] = *(const uint4*)(p + 16);
  return f.v;
}
// B fragment: matrix stored row-major as [col][k] ("B-transposed" storage),
// 32x16 tile. lane%16 = N col; per-lane elems 0..7 at k0+(lane/16)*8, elems 8..15 at +16.
__device__ __forceinline__ bf16x16 loadB(const u16* base, int ld, int col, int k0, int lane) {
  const u16* p = base + (size_t)col * ld + k0 + ((lane >> 4) << 3);
  FragU f;
  f.q[0] = *(const uint4*)(p);
  f.q[1] = *(const uint4*)(p + 16);
  return f.v;
}

#define WMMA_BF16(A, B, C) \
  __builtin_amdgcn_wmma_f32_16x16x32_bf16(false, (A), false, (B), (short)0, (C), false, false)

// ---------------- fp32 -> bf16 convert ----------------
__global__ void k_cvt(const float* __restrict__ in, u16* __restrict__ out, int n) {
  int i = blockIdx.x * blockDim.x + threadIdx.x;
  if (i < n) out[i] = f2bf(in[i]);
}

// ---------------- QKV GEMM: C[m,g] = sum_f X[m,f]*Wq[g,f] ----------------
// 32(M) x 64(N) tile per wave: 2 A frags reused across 4 B frags -> 8 WMMAs/K-step.
// q/k part -> QKVqk[m][0..1535] bf16 ; v part -> vT[b,h,e,n] bf16 (transposed)
__global__ void k_gemm_qkv(const u16* __restrict__ Xb, const u16* __restrict__ Wqb,
                           u16* __restrict__ qkout, u16* __restrict__ vT) {
  int lane = threadIdx.x & 31;
  int w    = threadIdx.x >> 5;
  int r    = lane & 15, hh = lane >> 4;
  int m0   = blockIdx.x << 5;
  int g0   = blockIdx.y * 256 + w * 64;
  f32x8 z = {0.f,0.f,0.f,0.f,0.f,0.f,0.f,0.f};
  f32x8 acc[2][4] = {{z, z, z, z}, {z, z, z, z}};
  for (int kk = 0; kk < DIMF; kk += 32) {
    if (kk + 32 < DIMF) {        // pull next K-step A cachelines (global_prefetch_b8)
      __builtin_prefetch(Xb + (size_t)(m0 + r) * DIMF + kk + 32, 0, 3);
      __builtin_prefetch(Xb + (size_t)(m0 + 16 + r) * DIMF + kk + 32, 0, 3);
    }
    bf16x16 a0 = loadA(Xb, DIMF, m0 + r,      kk, lane);
    bf16x16 a1 = loadA(Xb, DIMF, m0 + 16 + r, kk, lane);
#pragma unroll
    for (int t = 0; t < 4; ++t) {
      bf16x16 b = loadB(Wqb, DIMF, g0 + t * 16 + r, kk, lane);
      acc[0][t] = WMMA_BF16(a0, b, acc[0][t]);
      acc[1][t] = WMMA_BF16(a1, b, acc[1][t]);
    }
  }
  if (g0 < GQK) {                                   // uniform per wave
#pragma unroll
    for (int ms = 0; ms < 2; ++ms) {
#pragma unroll
      for (int t = 0; t < 4; ++t) {
        int g = g0 + t * 16 + r;
#pragma unroll
        for (int rr = 0; rr < 8; ++rr) {
          int m = m0 + ms * 16 + rr + (hh << 3);
          qkout[(size_t)m * GQK + g] = f2bf(acc[ms][t][rr]);
        }
      }
    }
  } else {
#pragma unroll
    for (int ms = 0; ms < 2; ++ms) {
#pragma unroll
      for (int t = 0; t < 4; ++t) {
        int e = g0 + t * 16 + r - GQK;
        int h = e >> 6, d = e & 63;
#pragma unroll
        for (int rr = 0; rr < 8; ++rr) {
          int m = m0 + ms * 16 + rr + (hh << 3);
          int b = m / NN, n = m - b * NN;
          vT[((size_t)(b * HH_ + h) * 64 + d) * NP + n] = f2bf(acc[ms][t][rr]);
        }
      }
    }
  }
}

// ------------- RoPE + d^-0.25 scale; zero-fill row pads; zero vT col pads -------------
__global__ void k_rope(const u16* __restrict__ qk, const float* __restrict__ fcos,
                       const float* __restrict__ fsin,
                       u16* __restrict__ qs, u16* __restrict__ ks, u16* __restrict__ vT) {
  int idx = blockIdx.x * blockDim.x + threadIdx.x;
  if (idx >= BH * NP * 32) return;
  int i  = idx & 31;
  int n  = (idx >> 5) % NP;
  int bh = idx / (NP * 32);
  size_t orow = ((size_t)bh * NP + n) * 64;
  if (n >= NN) {
    qs[orow + 2 * i] = 0; qs[orow + 2 * i + 1] = 0;
    ks[orow + 2 * i] = 0; ks[orow + 2 * i + 1] = 0;
    size_t vbase = (size_t)bh * 64 * NP + n;
    vT[vbase + (size_t)(2 * i) * NP]     = 0;
    vT[vbase + (size_t)(2 * i + 1) * NP] = 0;
    return;
  }
  int b = bh / HH_, h = bh % HH_;
  size_t m = (size_t)b * NN + n;
  const u16* qrow = qk + m * GQK + h * 64;
  const u16* krow = qrow + HD;
  float qe = bf2f(qrow[2 * i]), qo = bf2f(qrow[2 * i + 1]);
  float ke = bf2f(krow[2 * i]), ko = bf2f(krow[2 * i + 1]);
  if (n > 0) {
    float c = fcos[(size_t)(n - 1) * 32 + i];
    float s = fsin[(size_t)(n - 1) * 32 + i];
    float t0 = qe * c - qo * s; qo = qe * s + qo * c; qe = t0;
    float t1 = ke * c - ko * s; ko = ke * s + ko * c; ke = t1;
  }
  const float scale = 0.3535533905932738f;   // 64^-0.25
  qs[orow + 2 * i]     = f2bf(qe * scale);
  qs[orow + 2 * i + 1] = f2bf(qo * scale);
  ks[orow + 2 * i]     = f2bf(ke * scale);
  ks[orow + 2 * i + 1] = f2bf(ko * scale);
}

// ------------- feature map GEMM: relu(src @ P^T)+1e-6 -------------
// kmode==0 -> row-major qf[bh,n,j] ; kmode==1 -> transposed kfT[bh,j,n]
__global__ void k_feature(const u16* __restrict__ src, const u16* __restrict__ Pb,
                          u16* __restrict__ outRM, u16* __restrict__ outTR, int kmode) {
  int lane = threadIdx.x & 31;
  int w    = threadIdx.x >> 5;
  int T    = blockIdx.x * 4 + w;
  if (T >= BH * 65) return;
  int bh = T / 65, nt = T % 65;
  int r = lane & 15, hh = lane >> 4;
  int m0 = bh * NP + nt * 16;
  f32x8 z = {0.f,0.f,0.f,0.f,0.f,0.f,0.f,0.f};
  f32x8 acc[4] = {z, z, z, z};
  for (int kk = 0; kk < 64; kk += 32) {
    bf16x16 a = loadA(src, 64, m0 + r, kk, lane);
#pragma unroll
    for (int t = 0; t < 4; ++t) {
      bf16x16 b = loadB(Pb, 64, t * 16 + r, kk, lane);
      acc[t] = WMMA_BF16(a, b, acc[t]);
    }
  }
  if (kmode == 0) {
#pragma unroll
    for (int t = 0; t < 4; ++t) {
      int j = t * 16 + r;
#pragma unroll
      for (int rr = 0; rr < 8; ++rr) {
        float v = fmaxf(acc[t][rr], 0.f) + 1e-6f;
        outRM[((size_t)m0 + rr + hh * 8) * 64 + j] = f2bf(v);
      }
    }
  } else {
#pragma unroll
    for (int t = 0; t < 4; ++t) {
      int j = t * 16 + r;
      Pack8 p;
#pragma unroll
      for (int rr = 0; rr < 8; ++rr)
        p.h[rr] = f2bf(fmaxf(acc[t][rr], 0.f) + 1e-6f);
      size_t addr = ((size_t)bh * 64 + j) * NP + nt * 16 + hh * 8;
      *(uint4*)(outTR + addr) = p.q;   // 8 consecutive n -> one b128 store
    }
  }
}

// ------------- k_sum[bh,j] = sum_{n<N} kfT[bh,j,n] -------------
__global__ void k_ksum(const u16* __restrict__ kfT, float* __restrict__ ksum) {
  int row  = blockIdx.x * 4 + (threadIdx.x >> 5);
  int lane = threadIdx.x & 31;
  if (row >= BH * 64) return;
  const u16* p = kfT + (size_t)row * NP;
  float s = 0.f;
  for (int n = lane; n < NN; n += 32) s += bf2f(p[n]);
#pragma unroll
  for (int off = 16; off > 0; off >>= 1) s += __shfl_xor(s, off, 32);
  if (lane == 0) ksum[row] = s;
}

// ------------- D_inv[bh,n] = 1 / (qf[bh,n,:] . ksum[bh,:]) -------------
__global__ void k_dinv(const u16* __restrict__ qf, const float* __restrict__ ksum,
                       float* __restrict__ Dinv) {
  int idx = blockIdx.x * blockDim.x + threadIdx.x;
  if (idx >= BH * NP) return;
  int bh = idx / NP;
  const u16* q = qf + (size_t)idx * 64;
  const float* kv = ksum + bh * 64;
  float s = 0.f;
#pragma unroll
  for (int j = 0; j < 64; ++j) s += bf2f(q[j]) * kv[j];
  Dinv[idx] = 1.0f / s;
}

// ------------- context: ctx[j,e] = sum_n kf[n,j] v[n,e] ; store ctxT[bh,e,j] -------------
__global__ void k_context(const u16* __restrict__ kfT, const u16* __restrict__ vT,
                          u16* __restrict__ ctxT) {
  int lane = threadIdx.x & 31;
  int w    = threadIdx.x >> 5;
  int bh   = blockIdx.x;
  int r = lane & 15, hh = lane >> 4;
  int j0 = w * 16;
  const u16* kfbh = kfT + (size_t)bh * 64 * NP;
  const u16* vbh  = vT  + (size_t)bh * 64 * NP;
  f32x8 z = {0.f,0.f,0.f,0.f,0.f,0.f,0.f,0.f};
  f32x8 acc[4] = {z, z, z, z};
  for (int kk = 0; kk < NP; kk += 32) {
    if (kk + 32 < NP) {
      __builtin_prefetch(kfbh + (size_t)(j0 + r) * NP + kk + 32, 0, 3);
      __builtin_prefetch(vbh + (size_t)(r) * NP + kk + 32, 0, 3);
    }
    bf16x16 a = loadA(kfbh, NP, j0 + r, kk, lane);   // A[j][n] = kfT
#pragma unroll
    for (int t = 0; t < 4; ++t) {
      bf16x16 b = loadB(vbh, NP, t * 16 + r, kk, lane);  // B[n][e] via vT[e][n]
      acc[t] = WMMA_BF16(a, b, acc[t]);
    }
  }
#pragma unroll
  for (int t = 0; t < 4; ++t) {
    int e = t * 16 + r;
    Pack8 p;
#pragma unroll
    for (int rr = 0; rr < 8; ++rr) p.h[rr] = f2bf(acc[t][rr]);
    size_t addr = ((size_t)bh * 64 + e) * 64 + j0 + hh * 8;
    *(uint4*)(ctxT + addr) = p.q;
  }
}

// ------------- combine: attn[n,(h,e)] = Dinv[n] * sum_j qf[n,j]*ctx[j,e] -------------
__global__ void k_combine(const u16* __restrict__ qf, const u16* __restrict__ ctxT,
                          const float* __restrict__ Dinv, u16* __restrict__ attnb) {
  int lane = threadIdx.x & 31;
  int w    = threadIdx.x >> 5;
  int T    = blockIdx.x * 4 + w;
  if (T >= BH * 65) return;
  int bh = T / 65, nt = T % 65;
  int r = lane & 15, hh = lane >> 4;
  int m0 = bh * NP + nt * 16;
  const u16* cbh = ctxT + (size_t)bh * 64 * 64;
  f32x8 z = {0.f,0.f,0.f,0.f,0.f,0.f,0.f,0.f};
  f32x8 acc[4] = {z, z, z, z};
  for (int kk = 0; kk < 64; kk += 32) {
    bf16x16 a = loadA(qf, 64, m0 + r, kk, lane);
#pragma unroll
    for (int t = 0; t < 4; ++t) {
      bf16x16 b = loadB(cbh, 64, t * 16 + r, kk, lane);  // B[j][e] via ctxT[e][j]
      acc[t] = WMMA_BF16(a, b, acc[t]);
    }
  }
  int b = bh / HH_, h = bh % HH_;
#pragma unroll
  for (int t = 0; t < 4; ++t) {
    int col = h * 64 + t * 16 + r;
#pragma unroll
    for (int rr = 0; rr < 8; ++rr) {
      int n = nt * 16 + hh * 8 + rr;
      if (n < NN) {
        float v = acc[t][rr] * Dinv[bh * NP + n];
        attnb[((size_t)b * NN + n) * HD + col] = f2bf(v);
      }
    }
  }
}

// ------------- output projection: out[m,f] = sum_g attn[m,g]*Wo[f,g] + b[f] -------------
// 32(M) x 64(N) tile per wave.
__global__ void k_gemm_out(const u16* __restrict__ attnb, const u16* __restrict__ Wob,
                           const float* __restrict__ bout, float* __restrict__ out) {
  int lane = threadIdx.x & 31;
  int w    = threadIdx.x >> 5;
  int r    = lane & 15, hh = lane >> 4;
  int m0   = blockIdx.x << 5;
  int f0   = blockIdx.y * 256 + w * 64;
  f32x8 z = {0.f,0.f,0.f,0.f,0.f,0.f,0.f,0.f};
  f32x8 acc[2][4] = {{z, z, z, z}, {z, z, z, z}};
  for (int kk = 0; kk < HD; kk += 32) {
    if (kk + 32 < HD) {
      __builtin_prefetch(attnb + (size_t)(m0 + r) * HD + kk + 32, 0, 3);
      __builtin_prefetch(attnb + (size_t)(m0 + 16 + r) * HD + kk + 32, 0, 3);
    }
    bf16x16 a0 = loadA(attnb, HD, m0 + r,      kk, lane);
    bf16x16 a1 = loadA(attnb, HD, m0 + 16 + r, kk, lane);
#pragma unroll
    for (int t = 0; t < 4; ++t) {
      bf16x16 b = loadB(Wob, HD, f0 + t * 16 + r, kk, lane);
      acc[0][t] = WMMA_BF16(a0, b, acc[0][t]);
      acc[1][t] = WMMA_BF16(a1, b, acc[1][t]);
    }
  }
#pragma unroll
  for (int ms = 0; ms < 2; ++ms) {
#pragma unroll
    for (int t = 0; t < 4; ++t) {
      int f = f0 + t * 16 + r;
      float bias = bout[f];
#pragma unroll
      for (int rr = 0; rr < 8; ++rr) {
        int m = m0 + ms * 16 + rr + (hh << 3);
        out[(size_t)m * DIMF + f] = acc[ms][t][rr] + bias;
      }
    }
  }
}

extern "C" void kernel_launch(void* const* d_in, const int* in_sizes, int n_in,
                              void* d_out, int out_size, void* d_ws, size_t ws_size,
                              hipStream_t stream) {
  (void)in_sizes; (void)n_in; (void)out_size; (void)ws_size;
  const float* x  = (const float*)d_in[0];
  const float* Wq = (const float*)d_in[1];
  const float* Wo = (const float*)d_in[2];
  const float* bo = (const float*)d_in[3];
  const float* P  = (const float*)d_in[4];
  const float* fc = (const float*)d_in[5];
  const float* fs = (const float*)d_in[6];
  float* out = (float*)d_out;

  char* ws = (char*)d_ws;
  auto take = [&](size_t elems, size_t esz) -> char* {
    char* p = ws;
    ws += (elems * esz + 255) & ~(size_t)255;
    return p;
  };
  u16*   Xb    = (u16*)  take((size_t)BN * DIMF, 2);
  u16*   Wqb   = (u16*)  take((size_t)G3 * DIMF, 2);
  u16*   Wob   = (u16*)  take((size_t)DIMF * HD, 2);
  u16*   Pb    = (u16*)  take(64 * 64, 2);
  u16*   QKVqk = (u16*)  take((size_t)BN * GQK, 2);
  u16*   vT    = (u16*)  take((size_t)BH * 64 * NP, 2);
  u16*   qs    = (u16*)  take((size_t)BH * NP * 64, 2);
  u16*   ks    = (u16*)  take((size_t)BH * NP * 64, 2);
  u16*   qf    = (u16*)  take((size_t)BH * NP * 64, 2);
  u16*   kfT   = (u16*)  take((size_t)BH * 64 * NP, 2);
  float* ksum  = (float*)take((size_t)BH * 64, 4);
  u16*   ctxT  = (u16*)  take((size_t)BH * 64 * 64, 2);
  float* Dinv  = (float*)take((size_t)BH * NP, 4);
  u16*   attnb = (u16*)  take((size_t)BN * HD, 2);

  k_cvt<<<(BN * DIMF + 255) / 256, 256, 0, stream>>>(x,  Xb,  BN * DIMF);
  k_cvt<<<(G3 * DIMF + 255) / 256, 256, 0, stream>>>(Wq, Wqb, G3 * DIMF);
  k_cvt<<<(DIMF * HD + 255) / 256, 256, 0, stream>>>(Wo, Wob, DIMF * HD);
  k_cvt<<<(64 * 64 + 255) / 256,   256, 0, stream>>>(P,  Pb,  64 * 64);

  k_gemm_qkv<<<dim3(BN / 32, G3 / 256), 128, 0, stream>>>(Xb, Wqb, QKVqk, vT);

  int ropeTot = BH * NP * 32;
  k_rope<<<(ropeTot + 255) / 256, 256, 0, stream>>>(QKVqk, fc, fs, qs, ks, vT);

  k_feature<<<(BH * 65) / 4, 128, 0, stream>>>(qs, Pb, qf, (u16*)nullptr, 0);
  k_feature<<<(BH * 65) / 4, 128, 0, stream>>>(ks, Pb, (u16*)nullptr, kfT, 1);

  k_ksum<<<(BH * 64) / 4, 128, 0, stream>>>(kfT, ksum);
  k_dinv<<<(BH * NP + 255) / 256, 256, 0, stream>>>(qf, ksum, Dinv);

  k_context<<<BH, 128, 0, stream>>>(kfT, vT, ctxT);
  k_combine<<<(BH * 65) / 4, 128, 0, stream>>>(qf, ctxT, Dinv, attnb);

  k_gemm_out<<<dim3(BN / 32, DIMF / 256), 128, 0, stream>>>(attnb, Wob, bo, out);
}
